// GCNeXt_31430570672684
// MI455X (gfx1250) — matrix operand order repaired and checked by reading, not compiled, not run on hardware
//
#include <hip/hip_runtime.h>

#define BB 8
#define CC 256
#define NN 2000
#define KK 10
#define WIDTH 128
#define NTILES 125   // NN/16

typedef __attribute__((ext_vector_type(16))) _Float16 v16h;
typedef __attribute__((ext_vector_type(8)))  _Float16 v8h;
typedef __attribute__((ext_vector_type(8)))  float    v8f;
typedef __attribute__((ext_vector_type(4)))  int      v4i;
typedef __attribute__((ext_vector_type(8)))  int      v8i;

__device__ __forceinline__ v16h cat8(v8h a, v8h b) {
  return __builtin_shufflevector(a, b, 0,1,2,3,4,5,6,7,8,9,10,11,12,13,14,15);
}

// A fragment: 16x32 f16, row-major source (stride lda). Lane holds row m0+(lane&15);
// halves 0..15 = K k0+(lane&16?16:0)..+15.
__device__ __forceinline__ v16h load_a_frag(const _Float16* base, int lda, int m0, int k0, int lane) {
  const _Float16* p = base + (size_t)(m0 + (lane & 15)) * lda + (k0 + ((lane & 16) ? 16 : 0));
  v8h lo = *(const v8h*)p;
  v8h hi = *(const v8h*)(p + 8);
  return cat8(lo, hi);
}

// B fragment: 32x16 f16, source stored as one contiguous K-row per column (stride ldb).
// Lane holds column n0+(lane&15); halves 0..15 = K k0+(lane&16?16:0)..+15.
__device__ __forceinline__ v16h load_b_frag(const _Float16* base, int ldb, int n0, int k0, int lane) {
  const _Float16* p = base + (size_t)(n0 + (lane & 15)) * ldb + (k0 + ((lane & 16) ? 16 : 0));
  v8h lo = *(const v8h*)p;
  v8h hi = *(const v8h*)(p + 8);
  return cat8(lo, hi);
}

__device__ __forceinline__ v8f wmma16(v16h a, v16h b, v8f c) {
  return __builtin_amdgcn_wmma_f32_16x16x32_f16(false, a, false, b, (short)0, c, false, false);
}

// Async 16B gather directly into LDS (ASYNCcnt-tracked, no VGPR staging).
__device__ __forceinline__ void async_ld_b128(unsigned lds_off, const void* g) {
  asm volatile("global_load_async_to_lds_b128 %0, %1, off" :: "v"(lds_off), "v"(g) : "memory");
}
__device__ __forceinline__ void wait_async() {
  asm volatile("s_wait_asynccnt 0x0" ::: "memory");
}

// TDM: contiguous 1-D tile copy global -> LDS (nbytes multiple of 8, <= 16b elem count).
// D# group0/group1 packed per CDNA5 ISA 8.3/8.4: count=1, type=2, data_size=8B.
__device__ __forceinline__ void tdm_copy_1d(unsigned lds_addr, const void* gsrc, int n8 /*8B units*/) {
  unsigned long long ga = (unsigned long long)(size_t)gsrc;
  v4i g0;
  g0[0] = 1;                                            // count=1, user mode
  g0[1] = (int)lds_addr;                                // lds_addr (bytes)
  g0[2] = (int)(unsigned)ga;                            // global_addr[31:0]
  g0[3] = (int)(((unsigned)(ga >> 32)) & 0x01FFFFFFu) | (int)0x80000000u; // ga[56:32], type=2
  v8i g1;
  g1[0] = 0x00030000;                                   // data_size = 3 (8B), no multicast
  g1[1] = (n8 & 0xFFFF) << 16;                          // tensor_dim0[15:0]
  g1[2] = ((n8 >> 16) & 0xFFFF) | (1 << 16);            // tensor_dim0[31:16], tensor_dim1 = 1
  g1[3] = (n8 & 0xFFFF) << 16;                          // tile_dim0 = n8
  g1[4] = 0;                                            // tile_dim1 = tile_dim2 = 0 (unused)
  g1[5] = n8;                                           // tensor_dim0_stride lo32
  g1[6] = 0;
  g1[7] = 0;
  asm volatile("tensor_load_to_lds %0, %1" :: "s"(g0), "s"(g1) : "memory");
}

// ---------------- K0: transpose x (B,C,N)->(B*N,C), split f32 -> f16 hi/lo, row norms
__global__ void __launch_bounds__(256) k_prep(const float* __restrict__ x,
                                              _Float16* __restrict__ xhi,
                                              _Float16* __restrict__ xlo,
                                              float* __restrict__ xx) {
  int row = blockIdx.x;               // b*NN + n
  int b = row / NN, n = row % NN;
  int c = threadIdx.x;
  float v = x[((size_t)b * CC + c) * NN + n];
  _Float16 h = (_Float16)v;
  xhi[(size_t)row * CC + c] = h;
  xlo[(size_t)row * CC + c] = (_Float16)(v - (float)h);
  __shared__ float red[256];
  red[c] = v * v;
  __syncthreads();
  for (int s = 128; s > 0; s >>= 1) {
    if (c < s) red[c] += red[c + s];
    __syncthreads();
  }
  if (c == 0) xx[row] = red[0];
}

// ---------------- K0b: convert GEMM weight matrices to f16
__global__ void k_wcvt(const float* __restrict__ tW1, const float* __restrict__ tW3,
                       const float* __restrict__ sW1, const float* __restrict__ sW3,
                       _Float16* __restrict__ tW1h, _Float16* __restrict__ tW3h,
                       _Float16* __restrict__ sW1h, _Float16* __restrict__ sW3h) {
  int i = blockIdx.x * blockDim.x + threadIdx.x;
  int stride = gridDim.x * blockDim.x;
  for (int j = i; j < WIDTH * CC; j += stride)     tW1h[j] = (_Float16)tW1[j];
  for (int j = i; j < CC * WIDTH; j += stride)     tW3h[j] = (_Float16)tW3[j];
  for (int j = i; j < WIDTH * 2 * CC; j += stride) sW1h[j] = (_Float16)sW1[j];
  for (int j = i; j < CC * WIDTH; j += stride)     sW3h[j] = (_Float16)sW3[j];
}

// ---------------- K1: fused temporal branch for a 16-point tile
__global__ void __launch_bounds__(256) k_temporal(
    const _Float16* __restrict__ xhi, const int* __restrict__ nbr,
    const _Float16* __restrict__ tW1h, const float* __restrict__ tb1,
    const float* __restrict__ tW2, const float* __restrict__ tb2,
    const _Float16* __restrict__ tW3h, const float* __restrict__ tb3,
    float* __restrict__ tout) {
  __shared__ __align__(64) _Float16 colbuf[48 * 256];   // gathered neighbor columns
  __shared__ __align__(64) _Float16 hbuf[48 * 128];     // relu(W1 @ col) per column
  __shared__ __align__(64) _Float16 ybuf[16 * 128];     // after grouped 1x3
  int blk = blockIdx.x;
  int b  = blk / NTILES;
  int n0 = (blk % NTILES) * 16;
  int tid = threadIdx.x, lane = tid & 31, wv = tid >> 5;

  // async gather of 48 columns (16 points x 3 taps): 48 x 512B = 1536 x 16B moves
  const int* nbase = nbr + (size_t)(b * NN + n0) * 3;
  for (int it = 0; it < 6; ++it) {
    int idx = tid + it * 256;
    int col = idx >> 5;                 // 0..47
    int chunk = idx & 31;               // 32 x 16B chunks per 512B row
    int r = nbase[col];                 // flat row into B*N
    async_ld_b128((unsigned)(size_t)&colbuf[col * 256 + chunk * 8],
                  xhi + (size_t)r * 256 + chunk * 8);
  }
  wait_async();
  __syncthreads();

  // GEMM1: tW1(128x256) @ cols(256x48); wave w owns M-tile w*16
  int m0 = wv * 16;
  for (int nt = 0; nt < 3; ++nt) {
    int cb = nt * 16;
    v8f acc = {};
    for (int kc = 0; kc < 8; ++kc) {
      v16h a  = load_a_frag(tW1h, 256, m0, kc * 32, lane);
      v16h bm = load_b_frag(colbuf, 256, cb, kc * 32, lane);
      acc = wmma16(a, bm, acc);
    }
    int col = cb + (lane & 15);
    int mb  = m0 + ((lane & 16) ? 8 : 0);
    v8h hh;
    for (int r = 0; r < 8; ++r) {
      float vv = acc[r] + tb1[mb + r];
      hh[r] = (_Float16)(vv > 0.f ? vv : 0.f);
    }
    *(v8h*)&hbuf[col * 128 + mb] = hh;
  }
  __syncthreads();

  // grouped 1x3 valid conv over the 3 taps (tiny, VALU)
  for (int it = 0; it < 8; ++it) {
    int lin = tid + it * 256;
    int ln = lin >> 7, o = lin & 127;
    int g4 = (o >> 2) << 2;
    float acc = tb2[o];
    for (int c = 0; c < 4; ++c)
      for (int t = 0; t < 3; ++t)
        acc += (float)hbuf[(ln * 3 + t) * 128 + g4 + c] * tW2[(o * 4 + c) * 3 + t];
    ybuf[ln * 128 + o] = (_Float16)(acc > 0.f ? acc : 0.f);
  }
  __syncthreads();

  // GEMM2: tW3(256x128) @ y(128x16); each wave does 2 M-tiles
  for (int half = 0; half < 2; ++half) {
    int mt = (wv + half * 8) * 16;
    v8f acc = {};
    for (int kc = 0; kc < 4; ++kc) {
      v16h a  = load_a_frag(tW3h, 128, mt, kc * 32, lane);
      v16h bm = load_b_frag(ybuf, 128, 0, kc * 32, lane);
      acc = wmma16(a, bm, acc);
    }
    int col = lane & 15;
    int mb  = mt + ((lane & 16) ? 8 : 0);
    size_t rowoff = (size_t)(b * NN + n0 + col) * 256;
    float4 f0 = {acc[0] + tb3[mb + 0], acc[1] + tb3[mb + 1],
                 acc[2] + tb3[mb + 2], acc[3] + tb3[mb + 3]};
    float4 f1 = {acc[4] + tb3[mb + 4], acc[5] + tb3[mb + 5],
                 acc[6] + tb3[mb + 6], acc[7] + tb3[mb + 7]};
    *(float4*)&tout[rowoff + mb] = f0;
    *(float4*)&tout[rowoff + mb + 4] = f1;
  }
}

// ---------------- K2: fused KNN (split-f16 WMMA distances + in-LDS top-K)
__global__ void __launch_bounds__(256) k_knn(
    const _Float16* __restrict__ xhi, const _Float16* __restrict__ xlo,
    const float* __restrict__ xx, int* __restrict__ knn) {
  __shared__ __align__(64) _Float16 qhi[16 * 256];
  __shared__ __align__(64) _Float16 qlo[16 * 256];
  __shared__ float qxx[16];
  __shared__ float pdbuf[8][16][16];
  __shared__ float lval[8][16][KK];
  __shared__ int   lidx[8][16][KK];
  int blk = blockIdx.x;
  int b  = blk / NTILES;
  int n0 = (blk % NTILES) * 16;
  int tid = threadIdx.x, lane = tid & 31, wv = tid >> 5;
  size_t rowbase = (size_t)b * NN;

  // TDM: query tile rows n0..n0+15 are contiguous (16 x 512B = 8KB each of hi/lo)
  if (wv == 0) {
    tdm_copy_1d((unsigned)(size_t)qhi, xhi + (rowbase + n0) * 256, 1024);
    tdm_copy_1d((unsigned)(size_t)qlo, xlo + (rowbase + n0) * 256, 1024);
    __builtin_amdgcn_s_wait_tensorcnt(0);
  }
  if (tid < 16) qxx[tid] = xx[rowbase + n0 + tid];
  for (int i = tid; i < 8 * 16 * KK; i += 256) {
    (&lval[0][0][0])[i] = -3.0e38f;
    (&lidx[0][0][0])[i] = 0x7fffffff;
  }
  __syncthreads();

  // wave w scans candidate tiles w, w+8, ...
  for (int jt = wv; jt < NTILES; jt += 8) {
    int jb = jt * 16;
    int j  = jb + (lane & 15);
    const _Float16* bh = xhi + (rowbase + j) * 256;
    const _Float16* bl = xlo + (rowbase + j) * 256;
    if (jt + 8 < NTILES)
      __builtin_prefetch(xhi + (rowbase + (jt + 8) * 16 + (lane & 15)) * 256, 0, 1);
    v8f acc = {};
    for (int kc = 0; kc < 8; ++kc) {
      int k0 = kc * 32;
      v16h ah = load_a_frag(qhi, 256, 0, k0, lane);
      v16h al = load_a_frag(qlo, 256, 0, k0, lane);
      const _Float16* pbh = bh + k0 + ((lane & 16) ? 16 : 0);
      const _Float16* pbl = bl + k0 + ((lane & 16) ? 16 : 0);
      v16h bhf = cat8(*(const v8h*)pbh, *(const v8h*)(pbh + 8));
      v16h blf = cat8(*(const v8h*)pbl, *(const v8h*)(pbl + 8));
      acc = wmma16(ah, bhf, acc);   // hi*hi
      acc = wmma16(ah, blf, acc);   // hi*lo
      acc = wmma16(al, bhf, acc);   // lo*hi  (~fp32-accurate inner product)
    }
    float cxx = xx[rowbase + j];
    int mhi = (lane & 16) ? 8 : 0;
    for (int r = 0; r < 8; ++r) {
      int m = r + mhi;
      pdbuf[wv][m][lane & 15] = 2.0f * acc[r] - qxx[m] - cxx;
    }
    asm volatile("s_wait_dscnt 0" ::: "memory");   // wave-local LDS RAW fence
    if (lane < 16) {                               // re-converges before next WMMA
      int q = lane;
      float* V = lval[wv][q];
      int*   I = lidx[wv][q];
      for (int jl = 0; jl < 16; ++jl) {
        float v = pdbuf[wv][q][jl];
        int id = jb + jl;
        if (v > V[KK - 1]) {                       // strict > : ties keep lower index
          int pos = KK - 1;
          while (pos > 0 && v > V[pos - 1]) {
            V[pos] = V[pos - 1]; I[pos] = I[pos - 1]; --pos;
          }
          V[pos] = v; I[pos] = id;
        }
      }
    }
  }
  __syncthreads();

  // merge 8 partial top-K lists per query, tie-break by lower index
  if (tid < 16) {
    int q = tid;
    for (int t = 0; t < KK; ++t) {
      float best = -3.1e38f; int bi = 0x7fffffff, bw = 0, bp = 0;
      for (int w = 0; w < 8; ++w)
        for (int p = 0; p < KK; ++p) {
          float v = lval[w][q][p]; int id = lidx[w][q][p];
          if (v > best || (v == best && id < bi)) { best = v; bi = id; bw = w; bp = p; }
        }
      knn[(rowbase + n0 + q) * KK + t] = bi;       // local n index within batch
      lval[bw][q][bp] = -3.2e38f;
    }
  }
}

// ---------------- K3: fused semantic MLP + max-over-k + final combine
__global__ void __launch_bounds__(256) k_semantic(
    const float* __restrict__ x, const _Float16* __restrict__ xhi,
    const int* __restrict__ knn,
    const _Float16* __restrict__ sW1h, const float* __restrict__ sb1,
    const float* __restrict__ sW2, const float* __restrict__ sb2,
    const _Float16* __restrict__ sW3h, const float* __restrict__ sb3,
    const float* __restrict__ tout, float* __restrict__ out) {
  __shared__ __align__(64) _Float16 colbuf[16 * 512];  // [nbr(256) | center(256)] per col
  __shared__ __align__(64) _Float16 h1buf[16 * 128];
  __shared__ __align__(64) _Float16 gbuf[16 * 128];
  int blk = blockIdx.x;
  int b  = blk / NTILES;
  int n0 = (blk % NTILES) * 16;
  int tid = threadIdx.x, lane = tid & 31, wv = tid >> 5;
  size_t rowbase = (size_t)b * NN;

  v8f vmax0, vmax1;
  for (int r = 0; r < 8; ++r) { vmax0[r] = -3.0e38f; vmax1[r] = -3.0e38f; }

  for (int k = 0; k < KK; ++k) {
    // async gather: 16 cols x 1KB ([neighbor row | center row]) = 1024 x 16B moves
    for (int it = 0; it < 4; ++it) {
      int idx = tid + it * 256;
      int cn  = idx >> 6;                 // 0..15
      int c16 = idx & 63;                 // 64 x 16B chunks per 1KB column
      const _Float16* src;
      if (c16 < 32) {
        int id = knn[(rowbase + n0 + cn) * KK + k];
        src = xhi + (rowbase + id) * 256 + c16 * 8;
      } else {
        src = xhi + (rowbase + n0 + cn) * 256 + (c16 - 32) * 8;
      }
      async_ld_b128((unsigned)(size_t)&colbuf[cn * 512 + c16 * 8], src);
    }
    wait_async();
    __syncthreads();

    // GEMM1: sW1(128x512) @ xf(512x16)
    {
      int m0 = wv * 16;
      v8f acc = {};
      for (int kc = 0; kc < 16; ++kc) {
        v16h a  = load_a_frag(sW1h, 512, m0, kc * 32, lane);
        v16h bm = load_b_frag(colbuf, 512, 0, kc * 32, lane);
        acc = wmma16(a, bm, acc);
      }
      int col = lane & 15;
      int mb  = m0 + ((lane & 16) ? 8 : 0);
      v8h hh;
      for (int r = 0; r < 8; ++r) {
        float vv = acc[r] + sb1[mb + r];
        hh[r] = (_Float16)(vv > 0.f ? vv : 0.f);
      }
      *(v8h*)&h1buf[col * 128 + mb] = hh;
    }
    __syncthreads();

    // grouped 1x1 (tiny, VALU)
    for (int it = 0; it < 8; ++it) {
      int lin = tid + it * 256;
      int cn = lin >> 7, o = lin & 127;
      int g4 = (o >> 2) << 2;
      float acc = sb2[o];
      for (int c = 0; c < 4; ++c)
        acc += (float)h1buf[cn * 128 + g4 + c] * sW2[o * 4 + c];
      gbuf[cn * 128 + o] = (_Float16)(acc > 0.f ? acc : 0.f);
    }
    __syncthreads();

    // GEMM2: sW3(256x128) @ g(128x16), running max over k in registers
    for (int half = 0; half < 2; ++half) {
      int mt = (wv + half * 8) * 16;
      v8f acc = {};
      for (int kc = 0; kc < 4; ++kc) {
        v16h a  = load_a_frag(sW3h, 128, mt, kc * 32, lane);
        v16h bm = load_b_frag(gbuf, 128, 0, kc * 32, lane);
        acc = wmma16(a, bm, acc);
      }
      if (half == 0) { for (int r = 0; r < 8; ++r) vmax0[r] = fmaxf(vmax0[r], acc[r]); }
      else           { for (int r = 0; r < 8; ++r) vmax1[r] = fmaxf(vmax1[r], acc[r]); }
    }
    __syncthreads();
  }

  // final: out = relu(x + tout + sout)   (tb3 already in tout; sb3 added here)
  for (int half = 0; half < 2; ++half) {
    int mt  = (wv + half * 8) * 16;
    int col = lane & 15;
    int mb  = mt + ((lane & 16) ? 8 : 0);
    int n   = n0 + col;
    for (int r = 0; r < 8; ++r) {
      int m = mb + r;
      float s = (half == 0 ? vmax0[r] : vmax1[r]) + sb3[m];
      float val = s + tout[(rowbase + n) * 256 + m] + x[((size_t)b * CC + m) * NN + n];
      out[((size_t)b * CC + m) * NN + n] = val > 0.f ? val : 0.f;
    }
  }
}

extern "C" void kernel_launch(void* const* d_in, const int* in_sizes, int n_in,
                              void* d_out, int out_size, void* d_ws, size_t ws_size,
                              hipStream_t stream) {
  (void)in_sizes; (void)n_in; (void)out_size; (void)ws_size;
  const float* x   = (const float*)d_in[0];
  const int*   nbr = (const int*)d_in[2];
  const float* tW1 = (const float*)d_in[3];
  const float* tb1 = (const float*)d_in[4];
  const float* tW2 = (const float*)d_in[5];
  const float* tb2 = (const float*)d_in[6];
  const float* tW3 = (const float*)d_in[7];
  const float* tb3 = (const float*)d_in[8];
  const float* sW1 = (const float*)d_in[9];
  const float* sb1 = (const float*)d_in[10];
  const float* sW2 = (const float*)d_in[11];
  const float* sb2 = (const float*)d_in[12];
  const float* sW3 = (const float*)d_in[13];
  const float* sb3 = (const float*)d_in[14];
  float* out = (float*)d_out;

  char* ws = (char*)d_ws;
  size_t off = 0;
  auto take = [&](size_t bytes) -> void* {
    void* p = ws + off;
    off += (bytes + 255) & ~(size_t)255;
    return p;
  };
  _Float16* xhi  = (_Float16*)take((size_t)BB * NN * CC * 2);
  _Float16* xlo  = (_Float16*)take((size_t)BB * NN * CC * 2);
  float*    xx   = (float*)   take((size_t)BB * NN * 4);
  _Float16* tW1h = (_Float16*)take((size_t)WIDTH * CC * 2);
  _Float16* tW3h = (_Float16*)take((size_t)CC * WIDTH * 2);
  _Float16* sW1h = (_Float16*)take((size_t)WIDTH * 2 * CC * 2);
  _Float16* sW3h = (_Float16*)take((size_t)CC * WIDTH * 2);
  int*      kidx = (int*)     take((size_t)BB * NN * KK * 4);
  float*    tout = (float*)   take((size_t)BB * NN * CC * 4);

  k_prep<<<BB * NN, 256, 0, stream>>>(x, xhi, xlo, xx);
  k_wcvt<<<64, 256, 0, stream>>>(tW1, tW3, sW1, sW3, tW1h, tW3h, sW1h, sW3h);
  k_temporal<<<BB * NTILES, 256, 0, stream>>>(xhi, nbr, tW1h, tb1, tW2, tb2, tW3h, tb3, tout);
  k_knn<<<BB * NTILES, 256, 0, stream>>>(xhi, xlo, xx, kidx);
  k_semantic<<<BB * NTILES, 256, 0, stream>>>(x, xhi, kidx, sW1h, sb1, sW2, sb2, sW3h, sb3, tout, out);
}
